// ts_Learner_79439715107177
// MI455X (gfx1250) — compile-verified, matrix operand order checked by reading, not run on hardware
//
#include <hip/hip_runtime.h>
#include <hip/hip_bf16.h>

typedef __attribute__((ext_vector_type(16))) _Float16 v16h;
typedef __attribute__((ext_vector_type(8)))  float    v8f;

#define BDIM 256
#define SDIM 512
#define FD   128   // F0=F1
#define HDIM 256
#define GDIM 1024  // 4*H

static __device__ __forceinline__ v8f wmma16(v16h a, v16h b, v8f c) {
    return __builtin_amdgcn_wmma_f32_16x16x32_f16(false, a, false, b, (short)0, c, false, false);
}

// Fast transcendentals: gfx1250 has native V_TANH_F32.
static __device__ __forceinline__ float ftanh(float x) {
#if __has_builtin(__builtin_amdgcn_tanhf)
    return __builtin_amdgcn_tanhf(x);
#else
    return tanhf(x);
#endif
}
static __device__ __forceinline__ float fsigmoid(float x) {
#if __has_builtin(__builtin_amdgcn_tanhf)
    return 0.5f * __builtin_amdgcn_tanhf(0.5f * x) + 0.5f;
#else
    return 1.0f / (1.0f + __expf(-x));
#endif
}

// ---------------------------------------------------------------------------
// Pack a weight matrix into B-fragment layout:
// frag = kt*(N/16)+nt ; element g = frag*512 + lane*16 + i  (halfs)
// B(k,n): k = kt*32 + (lane>>4)*8 + (i&7) + (i>>3)*16 ; n = nt*16 + (lane&15)
// transposed==1 -> src is [N,K] (torch-style [out,in]); else src is [K,N]
// ---------------------------------------------------------------------------
__global__ void pack_bfrag(const float* __restrict__ src, _Float16* __restrict__ dst,
                           int K, int N, int transposed) {
    int g = blockIdx.x * blockDim.x + threadIdx.x;
    if (g >= K * N) return;
    int i    = g & 15;
    int lane = (g >> 4) & 31;
    int frag = g >> 9;
    int NT = N >> 4;
    int nt = frag % NT, kt = frag / NT;
    int k = kt * 32 + ((lane >> 4) * 8) + (i & 7) + ((i >> 3) * 16);
    int n = nt * 16 + (lane & 15);
    float v = transposed ? src[n * K + k] : src[k * N + n];
    dst[g] = (_Float16)v;
}

// ---------------------------------------------------------------------------
// Feature kernel helpers
// ---------------------------------------------------------------------------
__device__ __forceinline__ v16h load_a_lds(const _Float16* hb, int kt, int lane) {
    int row = lane & 15;
    int kb  = (lane >> 4) * 8;
    const _Float16* src = hb + row * FD + kt * 32 + kb;
    v16h a;
#pragma unroll
    for (int i = 0; i < 8; i++) { a[i] = src[i]; a[8 + i] = src[16 + i]; }
    return a;
}

__device__ __forceinline__ void mm_layer(const v16h* afr, int nkt,
                                         const v16h* __restrict__ bf,
                                         const float* __restrict__ bias,
                                         float* fb, int lane) {
    int ln = lane & 15, lh = lane >> 4;
    for (int nt = 0; nt < 8; nt++) {
        float bv = bias[nt * 16 + ln];
        v8f acc;
#pragma unroll
        for (int r = 0; r < 8; r++) acc[r] = bv;
        for (int kt = 0; kt < nkt; kt++)
            acc = wmma16(afr[kt], bf[((kt << 3) + nt) * 32 + lane], acc);
#pragma unroll
        for (int r = 0; r < 8; r++) {
            float v = acc[r];
            v = v > 0.0f ? v : 0.0f;                         // ReLU
            fb[(r + lh * 8) * FD + nt * 16 + ln] = v;
        }
    }
}

__device__ __forceinline__ void ln_row(const float* fb, _Float16* hb, int lane) {
    int row = lane & 15, half = lane >> 4;
    const float* src = fb + row * FD + half * 64;
    float s = 0.0f;
    for (int j = 0; j < 64; j++) s += src[j];
    s += __shfl_xor(s, 16, 32);
    float mean = s * (1.0f / 128.0f);
    float v = 0.0f;
    for (int j = 0; j < 64; j++) { float d = src[j] - mean; v += d * d; }
    v += __shfl_xor(v, 16, 32);
    float rstd = rsqrtf(v * (1.0f / 128.0f) + 1e-5f);
    _Float16* dst = hb + row * FD + half * 64;
    for (int j = 0; j < 64; j++) dst[j] = (_Float16)((src[j] - mean) * rstd);
}

// ---------------------------------------------------------------------------
// Feature kernel: per 16-token tile (16 batch rows @ fixed s), compute both
// MLP branches with WMMA + LDS LayerNorm, emit concat features as A-fragments.
// Block = 128 threads = 4 waves, one tile per wave.
// ---------------------------------------------------------------------------
__global__ __launch_bounds__(128) void feat_kernel(
    const float* __restrict__ x_num, const int* __restrict__ x_state,
    const float* __restrict__ emb,
    const v16h* __restrict__ nw0f, const float* __restrict__ nb0,
    const v16h* __restrict__ nw1f, const float* __restrict__ nb1,
    const v16h* __restrict__ sw0f, const float* __restrict__ sb0,
    const v16h* __restrict__ sw1f, const float* __restrict__ sb1,
    v16h* __restrict__ xfrag) {
    __shared__ float    fbuf[4][16 * FD];
    __shared__ _Float16 hbuf[4][16 * FD];
    int lane = threadIdx.x & 31;
    int wave = threadIdx.x >> 5;
    int tile = blockIdx.x * 4 + wave;       // 0 .. S*16-1
    int s     = tile >> 4;
    int btile = tile & 15;
    int ln = lane & 15, lh = lane >> 4;
    float*    fb = fbuf[wave];
    _Float16* hb = hbuf[wave];
    int b = btile * 16 + ln;                // this lane's A-row (batch index)
    size_t tok = (size_t)b * SDIM + s;

    v16h afr[4];
    // ---- numeric branch, layer 0 (K=32)
    {
        const float* xp = x_num + tok * 32 + lh * 8;
        v16h a;
#pragma unroll
        for (int i = 0; i < 8; i++) { a[i] = (_Float16)xp[i]; a[8 + i] = (_Float16)xp[16 + i]; }
        afr[0] = a;
    }
    mm_layer(afr, 1, nw0f, nb0, fb, lane);  __syncthreads();
    ln_row(fb, hb, lane);                   __syncthreads();
    // ---- numeric layer 1 (K=128)
#pragma unroll
    for (int kt = 0; kt < 4; kt++) afr[kt] = load_a_lds(hb, kt, lane);
    __syncthreads();
    mm_layer(afr, 4, nw1f, nb1, fb, lane);  __syncthreads();
    ln_row(fb, hb, lane);                   __syncthreads();
    // emit numeric half (k-tiles 0..3 of LSTM input)
    {
        v16h* dst = xfrag + ((size_t)(s * 16 + btile) * 8) * 32 + lane;
#pragma unroll
        for (int kt = 0; kt < 4; kt++) dst[kt * 32] = load_a_lds(hb, kt, lane);
    }
    __syncthreads();

    // ---- state branch, layer 0 (K=64, embedding gather)
    {
        const int* st = x_state + tok * 16;
#pragma unroll
        for (int kt = 0; kt < 2; kt++) {
            v16h a;
#pragma unroll
            for (int i = 0; i < 8; i++) {
                int k  = kt * 32 + lh * 8 + i;
                int k2 = k + 16;
                a[i]     = (_Float16)emb[st[k  >> 2] * 4 + (k  & 3)];
                a[8 + i] = (_Float16)emb[st[k2 >> 2] * 4 + (k2 & 3)];
            }
            afr[kt] = a;
        }
    }
    mm_layer(afr, 2, sw0f, sb0, fb, lane);  __syncthreads();
    ln_row(fb, hb, lane);                   __syncthreads();
#pragma unroll
    for (int kt = 0; kt < 4; kt++) afr[kt] = load_a_lds(hb, kt, lane);
    __syncthreads();
    mm_layer(afr, 4, sw1f, sb1, fb, lane);  __syncthreads();
    ln_row(fb, hb, lane);                   __syncthreads();
    // emit state half (k-tiles 4..7)
    {
        v16h* dst = xfrag + ((size_t)(s * 16 + btile) * 8 + 4) * 32 + lane;
#pragma unroll
        for (int kt = 0; kt < 4; kt++) dst[kt * 32] = load_a_lds(hb, kt, lane);
    }
}

// ---------------------------------------------------------------------------
// gates_x precompute: gx[t,btile,nt] = (x @ W + Bg) tile, stored in WMMA
// accumulator layout (lane-contiguous v8f) so the recurrence can C-init with
// two b128 loads. Throughput GEMM: 8192 blocks spread over the whole chip.
// ---------------------------------------------------------------------------
__global__ __launch_bounds__(256) void xw_kernel(const v16h* __restrict__ xfrag,
                                                 const v16h* __restrict__ Wf,
                                                 const float* __restrict__ Bg,
                                                 v8f* __restrict__ gx) {
    int lane = threadIdx.x & 31;
    int wave = threadIdx.x >> 5;          // 0..7
    int tile = blockIdx.x;                // t*16 + btile, 0..8191
    int ln = lane & 15;

    v16h ax[8];
    const v16h* xp = xfrag + (size_t)tile * 8 * 32 + lane;
#pragma unroll
    for (int kt = 0; kt < 8; kt++) ax[kt] = xp[kt * 32];

    for (int j = 0; j < 8; j++) {
        int nt = wave * 8 + j;            // 0..63
        float bv = Bg[nt * 16 + ln];
        v8f acc;
#pragma unroll
        for (int r = 0; r < 8; r++) acc[r] = bv;
        for (int kt = 0; kt < 8; kt++)
            acc = wmma16(ax[kt], Wf[((size_t)kt * 64 + nt) * 32 + lane], acc);
        gx[((size_t)tile * 64 + nt) * 32 + lane] = acc;
    }
}

// ---------------------------------------------------------------------------
// Recurrence-only LSTM (big-workspace path): gates C-init comes from gx,
// only the h@U half runs on the critical path (64 WMMAs/wave/step).
// ---------------------------------------------------------------------------
__global__ __launch_bounds__(256) void lstm2_kernel(
    const v8f* __restrict__ gx, const v16h* __restrict__ Uf,
    float* __restrict__ hlast) {
    __shared__ _Float16 hbuf[2][16 * HDIM];
    int lane  = threadIdx.x & 31;
    int wave  = threadIdx.x >> 5;   // 0..7
    int btile = blockIdx.x;         // 0..15
    int ln = lane & 15, lh = lane >> 4;

    for (int i = threadIdx.x; i < 16 * HDIM; i += blockDim.x) hbuf[0][i] = (_Float16)0.0f;
    __syncthreads();

    v8f cst[2] = {};
    int p = 0;

    for (int t = 0; t < SDIM; t++) {
        v16h ah[8];
        const _Float16* hb = hbuf[p];
#pragma unroll
        for (int kt = 0; kt < 8; kt++) {
            const _Float16* src = hb + ln * HDIM + kt * 32 + lh * 8;
            v16h a;
#pragma unroll
            for (int i = 0; i < 8; i++) { a[i] = src[i]; a[8 + i] = src[16 + i]; }
            ah[kt] = a;
        }
        const v8f* gp = gx + ((size_t)(t * 16 + btile) * 64) * 32 + lane;
        _Float16* hn = hbuf[1 - p];
#pragma unroll
        for (int q = 0; q < 2; q++) {
            int tb = wave * 2 + q;
            v8f ai = gp[(tb +  0) * 32];
            v8f af = gp[(tb + 16) * 32];
            v8f ag = gp[(tb + 32) * 32];
            v8f ao = gp[(tb + 48) * 32];
            for (int kt = 0; kt < 8; kt++) {
                const v16h* ub = Uf + ((size_t)kt * 64) * 32 + lane;
                ai = wmma16(ah[kt], ub[(tb +  0) * 32], ai);
                af = wmma16(ah[kt], ub[(tb + 16) * 32], af);
                ag = wmma16(ah[kt], ub[(tb + 32) * 32], ag);
                ao = wmma16(ah[kt], ub[(tb + 48) * 32], ao);
            }
#pragma unroll
            for (int r = 0; r < 8; r++) {
                float cn = fsigmoid(af[r]) * cst[q][r] + fsigmoid(ai[r]) * ftanh(ag[r]);
                cst[q][r] = cn;
                float hv = fsigmoid(ao[r]) * ftanh(cn);
                int row = r + lh * 8;
                hn[row * HDIM + tb * 16 + ln] = (_Float16)hv;
                if (t == SDIM - 1)
                    hlast[(btile * 16 + row) * HDIM + tb * 16 + ln] = hv;
            }
        }
        __syncthreads();
        p ^= 1;
    }
}

// ---------------------------------------------------------------------------
// Fused LSTM (small-workspace fallback): both x@W and h@U per step.
// ---------------------------------------------------------------------------
__global__ __launch_bounds__(256) void lstm_kernel(
    const v16h* __restrict__ xfrag, const v16h* __restrict__ Wf,
    const v16h* __restrict__ Uf,    const float* __restrict__ Bg,
    float* __restrict__ hlast) {
    __shared__ _Float16 hbuf[2][16 * HDIM];
    int lane  = threadIdx.x & 31;
    int wave  = threadIdx.x >> 5;   // 0..7
    int btile = blockIdx.x;         // 0..15
    int ln = lane & 15, lh = lane >> 4;

    for (int i = threadIdx.x; i < 16 * HDIM; i += blockDim.x) hbuf[0][i] = (_Float16)0.0f;
    __syncthreads();

    float bi[2], bfv[2], bgv[2], bo[2];
#pragma unroll
    for (int q = 0; q < 2; q++) {
        int tb = wave * 2 + q;
        bi[q]  = Bg[(tb +  0) * 16 + ln];
        bfv[q] = Bg[(tb + 16) * 16 + ln];
        bgv[q] = Bg[(tb + 32) * 16 + ln];
        bo[q]  = Bg[(tb + 48) * 16 + ln];
    }
    v8f cst[2] = {};
    int p = 0;

    for (int t = 0; t < SDIM; t++) {
        v16h ax[8], ah[8];
        const v16h* xp = xfrag + ((size_t)(t * 16 + btile) * 8) * 32 + lane;
#pragma unroll
        for (int kt = 0; kt < 8; kt++) ax[kt] = xp[kt * 32];
        const _Float16* hb = hbuf[p];
#pragma unroll
        for (int kt = 0; kt < 8; kt++) {
            const _Float16* src = hb + ln * HDIM + kt * 32 + lh * 8;
            v16h a;
#pragma unroll
            for (int i = 0; i < 8; i++) { a[i] = src[i]; a[8 + i] = src[16 + i]; }
            ah[kt] = a;
        }
        _Float16* hn = hbuf[1 - p];
#pragma unroll
        for (int q = 0; q < 2; q++) {
            int tb = wave * 2 + q;
            v8f ai, af, ag, ao;
#pragma unroll
            for (int r = 0; r < 8; r++) { ai[r] = bi[q]; af[r] = bfv[q]; ag[r] = bgv[q]; ao[r] = bo[q]; }
            for (int kt = 0; kt < 8; kt++) {
                const v16h* wb = Wf + ((size_t)kt * 64) * 32 + lane;
                ai = wmma16(ax[kt], wb[(tb +  0) * 32], ai);
                af = wmma16(ax[kt], wb[(tb + 16) * 32], af);
                ag = wmma16(ax[kt], wb[(tb + 32) * 32], ag);
                ao = wmma16(ax[kt], wb[(tb + 48) * 32], ao);
            }
            for (int kt = 0; kt < 8; kt++) {
                const v16h* ub = Uf + ((size_t)kt * 64) * 32 + lane;
                ai = wmma16(ah[kt], ub[(tb +  0) * 32], ai);
                af = wmma16(ah[kt], ub[(tb + 16) * 32], af);
                ag = wmma16(ah[kt], ub[(tb + 32) * 32], ag);
                ao = wmma16(ah[kt], ub[(tb + 48) * 32], ao);
            }
#pragma unroll
            for (int r = 0; r < 8; r++) {
                float cn = fsigmoid(af[r]) * cst[q][r] + fsigmoid(ai[r]) * ftanh(ag[r]);
                cst[q][r] = cn;
                float hv = fsigmoid(ao[r]) * ftanh(cn);
                int row = r + lh * 8;
                hn[row * HDIM + tb * 16 + ln] = (_Float16)hv;
                if (t == SDIM - 1)
                    hlast[(btile * 16 + row) * HDIM + tb * 16 + ln] = hv;
            }
        }
        __syncthreads();
        p ^= 1;
    }
}

// ---------------------------------------------------------------------------
// Head: out[b,c] = h_last[b,:] . out_w[c,:] + out_b[c]
// ---------------------------------------------------------------------------
__global__ void head_kernel(const float* __restrict__ hlast,
                            const float* __restrict__ out_w,
                            const float* __restrict__ out_b,
                            float* __restrict__ out) {
    int g = blockIdx.x * blockDim.x + threadIdx.x;
    if (g >= BDIM * 10) return;
    int b = g / 10, c = g % 10;
    float s = out_b[c];
    for (int k = 0; k < HDIM; k++) s += hlast[b * HDIM + k] * out_w[c * HDIM + k];
    out[g] = s;
}

// ---------------------------------------------------------------------------
extern "C" void kernel_launch(void* const* d_in, const int* in_sizes, int n_in,
                              void* d_out, int out_size, void* d_ws, size_t ws_size,
                              hipStream_t stream) {
    const float* x_num   = (const float*)d_in[0];
    const int*   x_state = (const int*)  d_in[1];
    const float* nw0 = (const float*)d_in[2];  const float* nb0 = (const float*)d_in[3];
    const float* nw1 = (const float*)d_in[4];  const float* nb1 = (const float*)d_in[5];
    const float* emb = (const float*)d_in[6];
    const float* sw0 = (const float*)d_in[7];  const float* sb0 = (const float*)d_in[8];
    const float* sw1 = (const float*)d_in[9];  const float* sb1 = (const float*)d_in[10];
    const float* W   = (const float*)d_in[11];
    const float* U   = (const float*)d_in[12];
    const float* Bg  = (const float*)d_in[13];
    const float* out_w = (const float*)d_in[14];
    const float* out_b = (const float*)d_in[15];
    float* out = (float*)d_out;

    char* ws = (char*)d_ws;
    _Float16* nw0f = (_Float16*)(ws + 0);          // 4096 halfs
    _Float16* nw1f = (_Float16*)(ws + 8192);       // 16384 halfs
    _Float16* sw0f = (_Float16*)(ws + 40960);      // 8192 halfs
    _Float16* sw1f = (_Float16*)(ws + 57344);      // 16384 halfs
    _Float16* Wf   = (_Float16*)(ws + 90112);      // 262144 halfs
    _Float16* Uf   = (_Float16*)(ws + 614400);     // 262144 halfs
    const size_t OFF_XFRAG = (size_t)2 * 1024 * 1024;                 // 64 MB
    const size_t OFF_HLAST = OFF_XFRAG + (size_t)64 * 1024 * 1024;    // 256 KB
    const size_t OFF_GX    = OFF_HLAST + (size_t)2 * 1024 * 1024;     // 512 MB
    const size_t NEED_BIG  = OFF_GX + (size_t)512 * 1024 * 1024;
    v16h*  xfrag = (v16h*)(ws + OFF_XFRAG);
    float* hlast = (float*)(ws + OFF_HLAST);
    v8f*   gx    = (v8f*)(ws + OFF_GX);
    bool big = ws_size >= NEED_BIG;

    // pack weights to B-fragment f16 layout
    pack_bfrag<<<(32  * 128  + 255) / 256, 256, 0, stream>>>(nw0, nw0f, 32,  128,  1);
    pack_bfrag<<<(128 * 128  + 255) / 256, 256, 0, stream>>>(nw1, nw1f, 128, 128,  1);
    pack_bfrag<<<(64  * 128  + 255) / 256, 256, 0, stream>>>(sw0, sw0f, 64,  128,  1);
    pack_bfrag<<<(128 * 128  + 255) / 256, 256, 0, stream>>>(sw1, sw1f, 128, 128,  1);
    pack_bfrag<<<(256 * 1024 + 255) / 256, 256, 0, stream>>>(W,   Wf,   256, 1024, 0);
    pack_bfrag<<<(256 * 1024 + 255) / 256, 256, 0, stream>>>(U,   Uf,   256, 1024, 0);

    // feature MLPs -> fragment-layout f16 activations
    feat_kernel<<<(SDIM * 16) / 4, 128, 0, stream>>>(
        x_num, x_state, emb,
        (const v16h*)nw0f, nb0, (const v16h*)nw1f, nb1,
        (const v16h*)sw0f, sb0, (const v16h*)sw1f, sb1, xfrag);

    if (big) {
        // throughput GEMM for the input half of the gates, then the
        // recurrence runs only h@U on its critical path
        xw_kernel<<<SDIM * 16, 256, 0, stream>>>(xfrag, (const v16h*)Wf, Bg, gx);
        lstm2_kernel<<<16, 256, 0, stream>>>(gx, (const v16h*)Uf, hlast);
    } else {
        lstm_kernel<<<16, 256, 0, stream>>>(xfrag, (const v16h*)Wf, (const v16h*)Uf, Bg, hlast);
    }

    // output head
    head_kernel<<<(BDIM * 10 + 255) / 256, 256, 0, stream>>>(hlast, out_w, out_b, out);
}
